// CorefModel_33663953666781
// MI455X (gfx1250) — matrix operand order repaired and checked by reading, not compiled, not run on hardware
//
#include <hip/hip_runtime.h>
#include <hip/hip_bf16.h>
#include <math.h>

// ---------------- problem constants ----------------
static constexpr int Wn   = 4096;   // tokens
static constexpr int Dn   = 768;    // bert dim
static constexpr int Kn   = 768;    // spans
static constexpr int Cn   = 50;     // antecedents
static constexpr int SPAN = 2304;   // 3*D
static constexpr int PAIR = 6932;   // 3*SPAN + 20
static constexpr int Hn   = 1000;
static constexpr int HP   = 1024;   // padded H
static constexpr int W1LD = 6944;   // padded PAIR stride (mult of 32 elems)
static constexpr int Msim = Kn * Cn; // 38400

typedef __attribute__((ext_vector_type(16))) __bf16 v16bf;
typedef __attribute__((ext_vector_type(8)))  float  v8f;

union V16U { v16bf v; uint4 q[2]; };
union B8U  { uint4 q; __bf16 e[8]; };

// ---- CDNA5 async global->LDS copy (no VGPR staging, ASYNCcnt-tracked) ----
#define ASYNC_LDS 1

__device__ __forceinline__ void lds_copy16(void* ldsPtr, const void* gPtr) {
#if ASYNC_LDS
    // Generic pointers to __shared__ carry the LDS byte offset in the low 32 bits.
    unsigned int loff = (unsigned int)(uintptr_t)ldsPtr;
    asm volatile("global_load_async_to_lds_b128 %0, %1, off"
                 :: "v"(loff), "v"(gPtr) : "memory");
#else
    *(uint4*)ldsPtr = *(const uint4*)gPtr;
#endif
}
__device__ __forceinline__ void lds_copy_wait() {
#if ASYNC_LDS
#if defined(__has_builtin) && __has_builtin(__builtin_amdgcn_s_wait_asynccnt)
    __builtin_amdgcn_s_wait_asynccnt(0);
#else
    asm volatile("s_wait_asynccnt 0x0" ::: "memory");
#endif
#endif
}

__device__ __forceinline__ int bucket_dist(int d) {
    if (d <= 4) return d < 0 ? 0 : d;
    int l = 31 - __clz(d);          // floor(log2(d)), d >= 5
    l += 3;
    return l > 9 ? 9 : l;
}

// A operand: 16x32 bf16 tile from row-major LDS (stride ldk elems).
// lane m<16 holds row m, K 0..15; lane m+16 holds K 16..31.
__device__ __forceinline__ v16bf load_a_op(const __bf16* lds, int row0, int kk, int lane, int ldk) {
    int m = lane & 15, hi = (lane >> 4) * 16;
    const __bf16* p = lds + (row0 + m) * ldk + kk;
    V16U u;
    u.q[0] = *(const uint4*)(p + hi);
    u.q[1] = *(const uint4*)(p + 8 + hi);
    return u.v;
}
// B operand: 32x16 tile, B stored transposed (Bt row-major [N,K]).
// lane n<16 holds Bt[n, 0..15]; lane n+16 holds Bt[n, 16..31].
__device__ __forceinline__ v16bf load_b_op(const __bf16* lds, int n0, int kk, int lane, int ldk) {
    const __bf16* p = lds + (n0 + (lane & 15)) * ldk + kk + (lane >> 4) * 16;
    V16U u;
    u.q[0] = *(const uint4*)(p);
    u.q[1] = *(const uint4*)(p + 8);
    return u.v;
}

// ---------------- stage 0: hybrid + token attention ----------------
__global__ __launch_bounds__(256) void hybrid_attn_kernel(
    const float* md, const float* md1, const float* attn_w, const float* attn_b,
    float* hybrid, float* token_attn) {
    int t = blockIdx.x;
    __shared__ float red[256];
    float s = 0.f;
    for (int c = threadIdx.x; c < Dn; c += 256) {
        float h = 0.5f * (md[(size_t)t * Dn + c] + md1[(size_t)t * Dn + c]);
        hybrid[(size_t)t * Dn + c] = h;
        s += h * attn_w[c];
    }
    red[threadIdx.x] = s; __syncthreads();
    for (int o = 128; o > 0; o >>= 1) {
        if (threadIdx.x < o) red[threadIdx.x] += red[threadIdx.x + o];
        __syncthreads();
    }
    if (threadIdx.x == 0) token_attn[t] = red[0] + attn_b[0];
}

// ---------------- transpose fp32 -> bf16 (with zero padding) ----------------
// out[r][c] = in[c][r] for c<inR, r<inC; else 0.   out is [outR, outC].
__global__ __launch_bounds__(256) void transpose_bf16_kernel(
    const float* in, __bf16* out, int inR, int inC, int outR, int outC) {
    __shared__ float tile[32][33];
    int oc0 = blockIdx.x * 32, or0 = blockIdx.y * 32;
    int tx = threadIdx.x & 31, ty = threadIdx.x >> 5;
    for (int i = 0; i < 32; i += 8) {
        int ir = oc0 + ty + i, ic = or0 + tx;
        tile[ty + i][tx] = (ir < inR && ic < inC) ? in[(size_t)ir * inC + ic] : 0.f;
    }
    __syncthreads();
    for (int i = 0; i < 32; i += 8) {
        int r = or0 + ty + i, c = oc0 + tx;
        if (r < outR && c < outC) out[(size_t)r * outC + c] = (__bf16)tile[tx][ty + i];
    }
}

// ---------------- distance prior score + G_dist (incl. bias b1) ----------------
__global__ __launch_bounds__(256) void dist_kernel(
    const float* prior_emb, const float* dist_w, const float* dist_b,
    const float* top_dist_emb, const float* W1, const float* b1,
    float* dist_score, float* Gd) {
    if (threadIdx.x < 10) {
        float s = 0.f;
        for (int f = 0; f < 20; f++) s += prior_emb[threadIdx.x * 20 + f] * dist_w[f];
        dist_score[threadIdx.x] = s + dist_b[0];
    }
    for (int idx = threadIdx.x; idx < 10 * HP; idx += 256) {
        int b = idx >> 10, h = idx & (HP - 1);
        float s = 0.f;
        if (h < Hn) {
            s = b1[h];
            for (int f = 0; f < 20; f++)
                s += top_dist_emb[b * 20 + f] * W1[(size_t)(3 * SPAN + f) * Hn + h];
        }
        Gd[idx] = s;
    }
}

// ---------------- span embeddings (start|end|head-attn), bf16 out ----------------
__global__ __launch_bounds__(256) void span_kernel(
    const float* md, const float* hybrid, const float* token_attn,
    const int* starts, const int* ends, __bf16* span) {
    int k = blockIdx.x;
    int s = starts[k], e = ends[k];
    int width = e - s + 1;
    __shared__ float prob[32];
    if (threadIdx.x == 0) {
        float mx = -INFINITY;
        for (int t = 0; t < width; t++) mx = fmaxf(mx, token_attn[s + t]);
        float den = 0.f;
        for (int t = 0; t < width; t++) { float p = __expf(token_attn[s + t] - mx); prob[t] = p; den += p; }
        float inv = 1.f / den;
        for (int t = 0; t < width; t++) prob[t] *= inv;
    }
    __syncthreads();
    for (int c = threadIdx.x; c < Dn; c += 256) {
        span[(size_t)k * SPAN + c]           = (__bf16)md[(size_t)s * Dn + c];
        span[(size_t)k * SPAN + Dn + c]      = (__bf16)md[(size_t)e * Dn + c];
        float h = 0.f;
        for (int t = 0; t < width; t++) h += prob[t] * hybrid[(size_t)(s + t) * Dn + c];
        span[(size_t)k * SPAN + 2 * Dn + c]  = (__bf16)h;
    }
}

// ---------------- generic NT WMMA GEMM: C[M,N] = A[M,K] * Bt[N,K]^T (+bias) ----------------
// 64x128 block tile, 256 threads = 8 waves (4 M-subtiles x 2 N-groups of 64).
static constexpr int KT = 64;
__global__ __launch_bounds__(256) void gemm_nt_kernel(
    const __bf16* A, int lda, const __bf16* Bt, int ldb,
    float* Cf, __bf16* Cb, int ldc, int Kd, const float* bias) {
    __shared__ __align__(16) __bf16 sA[64 * KT];
    __shared__ __align__(16) __bf16 sB[128 * KT];
    int tid = threadIdx.x, lane = tid & 31, w = tid >> 5;
    int mw = w & 3, nw = w >> 2;
    int m0 = blockIdx.x * 64, n0 = blockIdx.y * 128;
    v8f acc[4] = {};
    for (int k0 = 0; k0 < Kd; k0 += KT) {
        #pragma unroll
        for (int ci = 0; ci < 2; ci++) {
            int id = tid + ci * 256;
            int r = id >> 3, c8 = (id & 7) * 8;
            lds_copy16(sA + r * KT + c8, A + (size_t)(m0 + r) * lda + k0 + c8);
        }
        #pragma unroll
        for (int ci = 0; ci < 4; ci++) {
            int id = tid + ci * 256;
            int r = id >> 3, c8 = (id & 7) * 8;
            lds_copy16(sB + r * KT + c8, Bt + (size_t)(n0 + r) * ldb + k0 + c8);
        }
        lds_copy_wait();
        __syncthreads();
        #pragma unroll
        for (int ks = 0; ks < KT; ks += 32) {
            v16bf a = load_a_op(sA, mw * 16, ks, lane, KT);
            #pragma unroll
            for (int s = 0; s < 4; s++) {
                v16bf b = load_b_op(sB, nw * 64 + s * 16, ks, lane, KT);
                acc[s] = __builtin_amdgcn_wmma_f32_16x16x32_bf16(
                    false, a, false, b, (short)0, acc[s], false, false);
            }
        }
        __syncthreads();
    }
    int colbase = n0 + nw * 64 + (lane & 15);
    int rowbase = m0 + mw * 16 + (lane >> 4) * 8;
    #pragma unroll
    for (int s = 0; s < 4; s++) {
        int col = colbase + s * 16;
        float bv = bias ? bias[col] : 0.f;
        #pragma unroll
        for (int v = 0; v < 8; v++) {
            float val = acc[s][v] + bv;
            size_t o = (size_t)(rowbase + v) * ldc + col;
            if (Cf) Cf[o] = val; else Cb[o] = (__bf16)val;
        }
    }
}

// ---------------- mask + prior + top-50 + output init ----------------
__global__ __launch_bounds__(256) void topk_kernel(
    const float* fast, const float* ms, const float* dist_score, const float* b2,
    int* top_idx, float* out) {
    int k = blockIdx.x;
    __shared__ float sc[Kn];
    __shared__ float rv[256];
    __shared__ int   ri[256];
    float msk = ms[k];
    for (int j = threadIdx.x; j < Kn; j += 256) {
        int d = k - j;
        sc[j] = (d >= 1)
              ? fast[(size_t)k * Kn + j] + msk + ms[j] + dist_score[bucket_dist(d)]
              : -INFINITY;
    }
    if (threadIdx.x == 0) out[k * (Cn + 1)] = 0.f;   // epsilon column
    __syncthreads();
    float b2v = b2[0];
    for (int c = 0; c < Cn; c++) {
        float bv = -INFINITY; int bi = 0x7fffffff;
        for (int j = threadIdx.x; j < Kn; j += 256) {
            float v = sc[j];
            if (v > bv || (v == bv && j < bi)) { bv = v; bi = j; }
        }
        rv[threadIdx.x] = bv; ri[threadIdx.x] = bi; __syncthreads();
        for (int o = 128; o > 0; o >>= 1) {
            if (threadIdx.x < o) {
                float v2 = rv[threadIdx.x + o]; int i2 = ri[threadIdx.x + o];
                if (v2 > rv[threadIdx.x] || (v2 == rv[threadIdx.x] && i2 < ri[threadIdx.x])) {
                    rv[threadIdx.x] = v2; ri[threadIdx.x] = i2;
                }
            }
            __syncthreads();
        }
        if (threadIdx.x == 0) {
            int bj = ri[0];
            top_idx[k * Cn + c] = bj;
            out[k * (Cn + 1) + 1 + c] = rv[0] + b2v;   // top_fast + b2; slow added later
            sc[bj] = -INFINITY;
        }
        __syncthreads();
    }
}

// ---------------- fused sim-GEMM + FFNN epilogue ----------------
// M = K*C rows (A built on the fly: span[k] (*) span[j]), N = HP, K = SPAN.
// Epilogue: pre = acc + Gt[k,h] + Ga[j,h] + Gd[bucket,h]; relu; dot w2; atomicAdd out.
__global__ __launch_bounds__(256) void sim_gemm_kernel(
    const __bf16* span, const __bf16* W1c, const int* top_idx,
    const float* Gt, const float* Ga, const float* Gd,
    const float* w2, float* out) {
    __shared__ __align__(16) __bf16 sA[64 * KT];
    __shared__ __align__(16) __bf16 sB[128 * KT];
    __shared__ int   sK[64], sJ[64], sBk[64];
    __shared__ float rowsum[64];
    int tid = threadIdx.x, lane = tid & 31, w = tid >> 5;
    int mw = w & 3, nw = w >> 2;
    int m0 = blockIdx.x * 64, n0 = blockIdx.y * 128;
    if (tid < 64) {
        int r = m0 + tid;
        int kk = r / Cn, cc = r - kk * Cn;
        int j = top_idx[kk * Cn + cc];
        sK[tid] = kk; sJ[tid] = j; sBk[tid] = bucket_dist(kk - j);
        rowsum[tid] = 0.f;
    }
    __syncthreads();
    v8f acc[4] = {};
    for (int k0 = 0; k0 < SPAN; k0 += KT) {
        // B tile: plain copy -> async DMA to LDS
        #pragma unroll
        for (int ci = 0; ci < 4; ci++) {
            int id = tid + ci * 256;
            int r = id >> 3, c8 = (id & 7) * 8;
            lds_copy16(sB + r * KT + c8, W1c + (size_t)(n0 + r) * W1LD + k0 + c8);
        }
        // A tile: computed (span[k] (*) span[j]) -> VGPR path
        #pragma unroll
        for (int ci = 0; ci < 2; ci++) {
            int id = tid + ci * 256;
            int r = id >> 3, c8 = (id & 7) * 8;
            B8U x, y, z;
            x.q = *(const uint4*)(span + (size_t)sK[r] * SPAN + k0 + c8);
            y.q = *(const uint4*)(span + (size_t)sJ[r] * SPAN + k0 + c8);
            #pragma unroll
            for (int t = 0; t < 8; t++) z.e[t] = (__bf16)((float)x.e[t] * (float)y.e[t]);
            *(uint4*)(sA + r * KT + c8) = z.q;
        }
        lds_copy_wait();
        __syncthreads();
        #pragma unroll
        for (int ks = 0; ks < KT; ks += 32) {
            v16bf a = load_a_op(sA, mw * 16, ks, lane, KT);
            #pragma unroll
            for (int s = 0; s < 4; s++) {
                v16bf b = load_b_op(sB, nw * 64 + s * 16, ks, lane, KT);
                acc[s] = __builtin_amdgcn_wmma_f32_16x16x32_bf16(
                    false, a, false, b, (short)0, acc[s], false, false);
            }
        }
        __syncthreads();
    }
    int colbase = n0 + nw * 64 + (lane & 15);
    int rowloc  = mw * 16 + (lane >> 4) * 8;
    #pragma unroll
    for (int s = 0; s < 4; s++) {
        int col = colbase + s * 16;           // h index
        if (col < Hn) {
            float w2v = w2[col];
            #pragma unroll
            for (int v = 0; v < 8; v++) {
                int rl = rowloc + v;
                float pre = acc[s][v]
                          + Gt[(size_t)sK[rl] * HP + col]
                          + Ga[(size_t)sJ[rl] * HP + col]
                          + Gd[(size_t)sBk[rl] * HP + col];
                if (pre > 0.f) atomicAdd(&rowsum[rl], pre * w2v);
            }
        }
    }
    __syncthreads();
    if (tid < 64) {
        int r = m0 + tid;
        int kk = r / Cn, cc = r - kk * Cn;
        atomicAdd(&out[kk * (Cn + 1) + 1 + cc], rowsum[tid]);
    }
}

// ---------------- launcher ----------------
extern "C" void kernel_launch(void* const* d_in, const int* in_sizes, int n_in,
                              void* d_out, int out_size, void* d_ws, size_t ws_size,
                              hipStream_t stream) {
    const float* md        = (const float*)d_in[0];
    const float* md1       = (const float*)d_in[1];
    const int*   starts    = (const int*)d_in[2];
    const int*   ends      = (const int*)d_in[3];
    const float* ms        = (const float*)d_in[4];
    const float* attn_w    = (const float*)d_in[5];
    const float* attn_b    = (const float*)d_in[6];
    const float* coarse_w  = (const float*)d_in[7];
    const float* coarse_b  = (const float*)d_in[8];
    const float* prior_emb = (const float*)d_in[9];
    const float* dist_w    = (const float*)d_in[10];
    const float* dist_b    = (const float*)d_in[11];
    const float* tdist_emb = (const float*)d_in[12];
    const float* W1        = (const float*)d_in[13];
    const float* b1        = (const float*)d_in[14];
    const float* w2        = (const float*)d_in[15];
    const float* b2        = (const float*)d_in[16];
    float* out = (float*)d_out;

    // workspace layout (256B aligned)
    size_t off = 0;
    auto carve = [&](size_t bytes) { size_t o = off; off = (off + bytes + 255) & ~(size_t)255; return o; };
    char* ws = (char*)d_ws;
    float*  hybrid     = (float*)(ws + carve((size_t)Wn * Dn * 4));
    float*  token_attn = (float*)(ws + carve((size_t)Wn * 4));
    __bf16* span_bf    = (__bf16*)(ws + carve((size_t)Kn * SPAN * 2));
    __bf16* src_bf     = (__bf16*)(ws + carve((size_t)Kn * SPAN * 2));
    __bf16* cwT        = (__bf16*)(ws + carve((size_t)SPAN * SPAN * 2));
    __bf16* w1T        = (__bf16*)(ws + carve((size_t)HP * W1LD * 2));
    float*  fast       = (float*)(ws + carve((size_t)Kn * Kn * 4));
    float*  dscore     = (float*)(ws + carve(16 * 4));
    float*  Gd         = (float*)(ws + carve((size_t)10 * HP * 4));
    int*    tidx       = (int*)(ws + carve((size_t)Kn * Cn * 4));
    float*  Gt         = (float*)(ws + carve((size_t)Kn * HP * 4));
    float*  Ga         = (float*)(ws + carve((size_t)Kn * HP * 4));

    // 1) hybrid doc + token attention logits
    hybrid_attn_kernel<<<Wn, 256, 0, stream>>>(md, md1, attn_w, attn_b, hybrid, token_attn);
    // 2) bf16 NT weight layouts
    transpose_bf16_kernel<<<dim3(SPAN / 32, SPAN / 32), 256, 0, stream>>>(
        coarse_w, cwT, SPAN, SPAN, SPAN, SPAN);
    transpose_bf16_kernel<<<dim3(W1LD / 32, HP / 32), 256, 0, stream>>>(
        W1, w1T, PAIR, Hn, HP, W1LD);
    // 3) distance prior score + G_dist (carries b1)
    dist_kernel<<<1, 256, 0, stream>>>(prior_emb, dist_w, dist_b, tdist_emb, W1, b1, dscore, Gd);
    // 4) span embeddings
    span_kernel<<<Kn, 256, 0, stream>>>(md, hybrid, token_attn, starts, ends, span_bf);
    // 5) src = span @ coarse_w + coarse_b (bf16 out)
    gemm_nt_kernel<<<dim3(Kn / 64, SPAN / 128), 256, 0, stream>>>(
        span_bf, SPAN, cwT, SPAN, nullptr, src_bf, SPAN, SPAN, coarse_b);
    // 6) fast = src @ span^T
    gemm_nt_kernel<<<dim3(Kn / 64, Kn / 128), 256, 0, stream>>>(
        src_bf, SPAN, span_bf, SPAN, fast, nullptr, Kn, SPAN, nullptr);
    // 7) mask + prior + top-50, init d_out with eps column and top_fast + b2
    topk_kernel<<<Kn, 256, 0, stream>>>(fast, ms, dscore, b2, tidx, out);
    // 8) G_tgt = span @ W1[0:SPAN], G_ant = span @ W1[SPAN:2*SPAN]
    gemm_nt_kernel<<<dim3(Kn / 64, HP / 128), 256, 0, stream>>>(
        span_bf, SPAN, w1T, W1LD, Gt, nullptr, HP, SPAN, nullptr);
    gemm_nt_kernel<<<dim3(Kn / 64, HP / 128), 256, 0, stream>>>(
        span_bf, SPAN, w1T + SPAN, W1LD, Ga, nullptr, HP, SPAN, nullptr);
    // 9) fused sim-GEMM + ReLU FFNN reduction into d_out
    sim_gemm_kernel<<<dim3(Msim / 64, HP / 128), 256, 0, stream>>>(
        span_bf, w1T + 2 * SPAN, tidx, Gt, Ga, Gd, w2, out);
}
